// TemplateEmbeddingIteration_36172214566960
// MI455X (gfx1250) — matrix operand order repaired and checked by reading, not compiled, not run on hardware
//
#include <hip/hip_runtime.h>
#include <hip/hip_bf16.h>

typedef _Float16 f16;
typedef __attribute__((ext_vector_type(16))) _Float16 v16h;
typedef __attribute__((ext_vector_type(8)))  float    v8f;

#define WPB   4        // waves per block for WMMA kernels (blockDim = 128)
#define NRES  256
#define CH    64
#define NPOS  65536    // NRES*NRES
#define NHEAD 4
#define ACH   16

// ---------------------------------------------------------------- WMMA utils

__device__ inline v8f wmma_f16(v16h a, v16h b, v8f c) {
  return __builtin_amdgcn_wmma_f32_16x16x32_f16(false, a, false, b, (short)0, c,
                                                false, false);
}

__device__ inline float sigmf(float x) { return 1.f / (1.f + __expf(-x)); }

// A fragment (16x32 f16, row-major MxK source). Lanes 0-15: K = k0..k0+15;
// lanes 16-31: K = k0+16..k0+31.
__device__ inline v16h a_frag(const f16* base, int ldk, int lane, int k0) {
  int m = lane & 15;
  int k = k0 + ((lane & 16) ? 16 : 0);
  const f16* p = base + (size_t)m * ldk + k;
  v16h r;
#pragma unroll
  for (int t = 0; t < 16; ++t) r[t] = p[t];
  return r;
}

// B fragment (32x16 f16) loaded from Bt = B^T, row-major NxK. Lane n holds
// column n; lanes 0-15: K = k0..k0+15, lanes 16-31: K = k0+16..k0+31.
__device__ inline v16h b_frag(const f16* bt, int ldk, int lane, int k0) {
  int n = lane & 15;
  int k = k0 + ((lane & 16) ? 16 : 0);
  const f16* p = bt + (size_t)n * ldk + k;
  v16h r;
#pragma unroll
  for (int t = 0; t < 16; ++t) r[t] = p[t];
  return r;
}

// Per-wave LayerNorm of 16 rows x 64ch f32 -> f16 into LDS (wave-private).
__device__ inline void ln_rows_to_lds(const float* src, const float* g,
                                      const float* b, f16* lds, int lane) {
  int row = lane >> 1, half = lane & 1;
  const float* p = src + (size_t)row * CH + half * 32;
  float vals[32];
  float s = 0.f;
#pragma unroll
  for (int t = 0; t < 32; ++t) { vals[t] = p[t]; s += vals[t]; }
  s += __shfl_xor(s, 1, 32);
  float mean = s * (1.f / 64.f);
  float vs = 0.f;
#pragma unroll
  for (int t = 0; t < 32; ++t) { float d = vals[t] - mean; vs += d * d; }
  vs += __shfl_xor(vs, 1, 32);
  float rstd = rsqrtf(vs * (1.f / 64.f) + 1e-5f);
#pragma unroll
  for (int t = 0; t < 32; ++t) {
    int c = half * 32 + t;
    lds[row * CH + c] = (f16)((vals[t] - mean) * rstd * g[c] + b[c]);
  }
}

#define WAVE_LDS_FENCE() asm volatile("s_wait_dscnt 0" ::: "memory")

// ---------------------------------------------------------------- prep

__global__ void convert_wt_kernel(const float* __restrict__ w, f16* __restrict__ wt,
                                  int kin, int nout) {
  int idx = blockIdx.x * 256 + threadIdx.x;
  if (idx >= kin * nout) return;
  int n = idx / kin, k = idx - n * kin;
  wt[idx] = (f16)w[(size_t)k * nout + n];   // wt[n][k] = w[k][n]
}

// w2d [64,4] -> padded f16 [16][64] (rows >=4 zero)
__global__ void convert_w2d_kernel(const float* __restrict__ w, f16* __restrict__ wt) {
  int idx = blockIdx.x * 256 + threadIdx.x;
  if (idx >= 16 * 64) return;
  int n = idx >> 6, k = idx & 63;
  wt[idx] = (n < NHEAD) ? (f16)w[(size_t)k * NHEAD + n] : (f16)0.f;
}

// ---------------------------------------------------------------- LN -> f16

__global__ __launch_bounds__(256)
void ln_to_f16_kernel(const float* __restrict__ src, const float* __restrict__ g,
                      const float* __restrict__ b, f16* __restrict__ dst,
                      int transpose) {
  int wave = threadIdx.x >> 5, lane = threadIdx.x & 31;
  int r = (blockIdx.x * 8 + wave) * 16 + (lane >> 1);
  int half = lane & 1;
  const float* p = src + (size_t)r * CH + half * 32;
  float vals[32];
  float s = 0.f;
#pragma unroll
  for (int t = 0; t < 32; ++t) { vals[t] = p[t]; s += vals[t]; }
  s += __shfl_xor(s, 1, 32);
  float mean = s * (1.f / 64.f);
  float vs = 0.f;
#pragma unroll
  for (int t = 0; t < 32; ++t) { float d = vals[t] - mean; vs += d * d; }
  vs += __shfl_xor(vs, 1, 32);
  float rstd = rsqrtf(vs * (1.f / 64.f) + 1e-5f);
  int pd = transpose ? (((r & 255) << 8) | (r >> 8)) : r;
  f16* q = dst + (size_t)pd * CH + half * 32;
#pragma unroll
  for (int t = 0; t < 32; ++t) {
    int c = half * 32 + t;
    q[t] = (f16)((vals[t] - mean) * rstd * g[c] + b[c]);
  }
}

// ------------------------------------------------ triangle mult: projections

__global__ __launch_bounds__(128)
void trimul_proj_kernel(const f16* __restrict__ xln,
                        const f16* wlT, const f16* wlgT, const f16* wrT,
                        const f16* wrgT, const f16* wgT,
                        const float* bl, const float* blg, const float* br,
                        const float* brg, const float* bg,
                        const float* __restrict__ mask,
                        f16* __restrict__ L, f16* __restrict__ R,
                        float* __restrict__ gate, int transpose) {
  int wave = threadIdx.x >> 5, lane = threadIdx.x & 31;
  int r0 = (blockIdx.x * WPB + wave) * 16;
  const f16* arow = xln + (size_t)r0 * CH;
  v16h a0 = a_frag(arow, CH, lane, 0);
  v16h a1 = a_frag(arow, CH, lane, 32);
  int n = lane & 15, roff = (lane & 16) ? 8 : 0;
#pragma unroll
  for (int ct = 0; ct < 4; ++ct) {
    int c = ct * 16 + n;
    const int wo = ct * 16 * CH;
    v8f al = {}, alg = {}, ar = {}, arg_ = {}, ag = {};
    al   = wmma_f16(a0, b_frag(wlT  + wo, CH, lane, 0),  al);
    al   = wmma_f16(a1, b_frag(wlT  + wo, CH, lane, 32), al);
    alg  = wmma_f16(a0, b_frag(wlgT + wo, CH, lane, 0),  alg);
    alg  = wmma_f16(a1, b_frag(wlgT + wo, CH, lane, 32), alg);
    ar   = wmma_f16(a0, b_frag(wrT  + wo, CH, lane, 0),  ar);
    ar   = wmma_f16(a1, b_frag(wrT  + wo, CH, lane, 32), ar);
    arg_ = wmma_f16(a0, b_frag(wrgT + wo, CH, lane, 0),  arg_);
    arg_ = wmma_f16(a1, b_frag(wrgT + wo, CH, lane, 32), arg_);
    ag   = wmma_f16(a0, b_frag(wgT  + wo, CH, lane, 0),  ag);
    ag   = wmma_f16(a1, b_frag(wgT  + wo, CH, lane, 32), ag);
    float vbl = bl[c], vblg = blg[c], vbr = br[c], vbrg = brg[c], vbg = bg[c];
#pragma unroll
    for (int v = 0; v < 8; ++v) {
      int r = r0 + roff + v;
      float mk = mask[r];
      int i = r >> 8, j = r & 255;
      size_t pos = transpose ? ((size_t)j * NRES + i) : ((size_t)i * NRES + j);
      L[(size_t)c * NPOS + pos] = (f16)(mk * (al[v] + vbl) * sigmf(alg[v] + vblg));
      R[(size_t)c * NPOS + pos] = (f16)(mk * (ar[v] + vbr) * sigmf(arg_[v] + vbrg));
      gate[(size_t)r * CH + c]  = sigmf(ag[v] + vbg);
    }
  }
}

// --------------------------- triangle einsum: out[i,j,c] = sum_k A[c,i,k]B[c,j,k]

__global__ __launch_bounds__(128)
void trimul_gemm_kernel(const f16* __restrict__ A, const f16* __restrict__ B,
                        float* __restrict__ out) {
  int wave = threadIdx.x >> 5, lane = threadIdx.x & 31;
  int wid = blockIdx.x * WPB + wave;       // 4096 waves: 64 ch x 8 x 8 (32x32 blocks)
  int c = wid >> 6;
  int t = wid & 63;
  int bi = (t >> 3) * 32, bj = (t & 7) * 32;
  const f16* Ac = A + (size_t)c * NPOS;
  const f16* Bc = B + (size_t)c * NPOS;
  v8f z = {};
  v8f acc00 = z, acc01 = z, acc10 = z, acc11 = z;
  for (int k0 = 0; k0 < NRES; k0 += 32) {
    v16h a0 = a_frag(Ac + (size_t)bi * NRES, NRES, lane, k0);
    v16h a1 = a_frag(Ac + (size_t)(bi + 16) * NRES, NRES, lane, k0);
    v16h b0 = b_frag(Bc + (size_t)bj * NRES, NRES, lane, k0);
    v16h b1 = b_frag(Bc + (size_t)(bj + 16) * NRES, NRES, lane, k0);
    acc00 = wmma_f16(a0, b0, acc00);
    acc01 = wmma_f16(a0, b1, acc01);
    acc10 = wmma_f16(a1, b0, acc10);
    acc11 = wmma_f16(a1, b1, acc11);
  }
  int n = lane & 15, roff = (lane & 16) ? 8 : 0;
#pragma unroll
  for (int v = 0; v < 8; ++v) {
    int i0 = bi + roff + v, i1 = i0 + 16;
    int j0 = bj + n, j1 = j0 + 16;
    out[((size_t)i0 * NRES + j0) * CH + c] = acc00[v];
    out[((size_t)i0 * NRES + j1) * CH + c] = acc01[v];
    out[((size_t)i1 * NRES + j0) * CH + c] = acc10[v];
    out[((size_t)i1 * NRES + j1) * CH + c] = acc11[v];
  }
}

// ---------------------- triangle mult epilogue: cur += gate * (cLN(tri)@wo+bo)

__global__ __launch_bounds__(128)
void trimul_fuse_kernel(const float* __restrict__ tri, const float* cg,
                        const float* cb, const f16* __restrict__ woT,
                        const float* bo, const float* __restrict__ gate,
                        float* __restrict__ cur) {
  __shared__ __align__(16) f16 lds[WPB][16 * CH];
  int wave = threadIdx.x >> 5, lane = threadIdx.x & 31;
  int r0 = (blockIdx.x * WPB + wave) * 16;
  f16* l = lds[wave];
  ln_rows_to_lds(tri + (size_t)r0 * CH, cg, cb, l, lane);
  WAVE_LDS_FENCE();
  v16h a0 = a_frag(l, CH, lane, 0);
  v16h a1 = a_frag(l, CH, lane, 32);
  int n = lane & 15, roff = (lane & 16) ? 8 : 0;
#pragma unroll
  for (int ct = 0; ct < 4; ++ct) {
    v8f acc = {};
    acc = wmma_f16(a0, b_frag(woT + ct * 16 * CH, CH, lane, 0), acc);
    acc = wmma_f16(a1, b_frag(woT + ct * 16 * CH, CH, lane, 32), acc);
    int c = ct * 16 + n;
    float vbo = bo[c];
#pragma unroll
    for (int v = 0; v < 8; ++v) {
      size_t idx = (size_t)(r0 + roff + v) * CH + c;
      cur[idx] += (acc[v] + vbo) * gate[idx];
    }
  }
}

// ---------------------------------------------------- attention: projections

__global__ __launch_bounds__(128)
void att_proj_kernel(const f16* __restrict__ xln, const f16* wqT, const f16* wkT,
                     const f16* wvT, const f16* wgT, const f16* w2dT,
                     const float* bg, f16* __restrict__ q, f16* __restrict__ k,
                     f16* __restrict__ vt, float* __restrict__ g,
                     float* __restrict__ nb) {
  int wave = threadIdx.x >> 5, lane = threadIdx.x & 31;
  int r0 = (blockIdx.x * WPB + wave) * 16;
  const f16* arow = xln + (size_t)r0 * CH;
  v16h a0 = a_frag(arow, CH, lane, 0);
  v16h a1 = a_frag(arow, CH, lane, 32);
  int n = lane & 15, roff = (lane & 16) ? 8 : 0;
#pragma unroll
  for (int ct = 0; ct < 4; ++ct) {          // ct == head (C=64, H=4, ACH=16)
    const int wo = ct * 16 * CH;
    int h = ct, c = ct * 16 + n;
    {
      v8f acc = {};
      acc = wmma_f16(a0, b_frag(wqT + wo, CH, lane, 0), acc);
      acc = wmma_f16(a1, b_frag(wqT + wo, CH, lane, 32), acc);
#pragma unroll
      for (int v = 0; v < 8; ++v) {
        int r = r0 + roff + v; int i = r >> 8, j = r & 255;
        q[(((size_t)i * NHEAD + h) * NRES + j) * ACH + n] = (f16)(acc[v] * 0.25f);
      }
    }
    {
      v8f acc = {};
      acc = wmma_f16(a0, b_frag(wkT + wo, CH, lane, 0), acc);
      acc = wmma_f16(a1, b_frag(wkT + wo, CH, lane, 32), acc);
#pragma unroll
      for (int v = 0; v < 8; ++v) {
        int r = r0 + roff + v; int i = r >> 8, j = r & 255;
        k[(((size_t)i * NHEAD + h) * NRES + j) * ACH + n] = (f16)acc[v];
      }
    }
    {
      v8f acc = {};
      acc = wmma_f16(a0, b_frag(wvT + wo, CH, lane, 0), acc);
      acc = wmma_f16(a1, b_frag(wvT + wo, CH, lane, 32), acc);
#pragma unroll
      for (int v = 0; v < 8; ++v) {
        int r = r0 + roff + v; int i = r >> 8, j = r & 255;
        vt[(((size_t)i * NHEAD + h) * ACH + n) * NRES + j] = (f16)acc[v];
      }
    }
    {
      v8f acc = {};
      acc = wmma_f16(a0, b_frag(wgT + wo, CH, lane, 0), acc);
      acc = wmma_f16(a1, b_frag(wgT + wo, CH, lane, 32), acc);
      float vbg = bg[c];
#pragma unroll
      for (int v = 0; v < 8; ++v)
        g[(size_t)(r0 + roff + v) * CH + c] = sigmf(acc[v] + vbg);
    }
  }
  {
    v8f acc = {};
    acc = wmma_f16(a0, b_frag(w2dT, CH, lane, 0), acc);
    acc = wmma_f16(a1, b_frag(w2dT, CH, lane, 32), acc);
    if (n < NHEAD) {
#pragma unroll
      for (int v = 0; v < 8; ++v)
        nb[(size_t)n * NPOS + (r0 + roff + v)] = acc[v];
    }
  }
}

// --------------------------------------- attention core (flash softmax rows)

__global__ __launch_bounds__(128)
void att_core_kernel(const f16* __restrict__ q, const f16* __restrict__ k,
                     const f16* __restrict__ vt, const float* __restrict__ nb,
                     const float* __restrict__ mask, float* __restrict__ o,
                     int tmask) {
  __shared__ __align__(16) f16 plds[WPB][256];
  int wave = threadIdx.x >> 5, lane = threadIdx.x & 31;
  int wid = blockIdx.x * WPB + wave;     // 16384 waves: i(256) x h(4) x jt(16)
  int jt = wid & 15, h = (wid >> 4) & 3, i = wid >> 6;
  int n = lane & 15, roff = (lane & 16) ? 8 : 0;

  const f16* qb = q + (((size_t)i * NHEAD + h) * NRES + jt * 16) * ACH;
  const f16* kb = k + ((size_t)i * NHEAD + h) * NRES * ACH;
  const f16* vb = vt + ((size_t)i * NHEAD + h) * ACH * NRES;
  const float* nbh = nb + (size_t)h * NPOS;
  f16* pl = plds[wave];

  // A fragment: Q rows, K = ACH = 16, upper half zero-padded.
  v16h af;
#pragma unroll
  for (int t = 0; t < 16; ++t) af[t] = (f16)0.f;
  if (!(lane & 16)) {
    int m = lane & 15;
    const f16* p = qb + m * ACH;
#pragma unroll
    for (int t = 0; t < 16; ++t) af[t] = p[t];
  }

  float rm[8], rs[8];
#pragma unroll
  for (int v = 0; v < 8; ++v) { rm[v] = -1e30f; rs[v] = 0.f; }
  v8f oacc = {};

  for (int kt = 0; kt < 16; ++kt) {
    int kcol = kt * 16 + n;
    // B fragment from K rows (K^T), K = 16 zero-padded.
    v16h bf;
#pragma unroll
    for (int t = 0; t < 16; ++t) bf[t] = (f16)0.f;
    if (!(lane & 16)) {
      const f16* p = kb + (size_t)(kt * 16 + n) * ACH;
#pragma unroll
      for (int t = 0; t < 16; ++t) bf[t] = p[t];
    }
    v8f z = {};
    v8f s = wmma_f16(af, bf, z);

    float mbias = tmask ? 1e9f * (mask[(size_t)kcol * NRES + i] - 1.f)
                        : 1e9f * (mask[(size_t)i * NRES + kcol] - 1.f);
#pragma unroll
    for (int v = 0; v < 8; ++v) {
      int j = jt * 16 + roff + v;
      float sv = s[v] + mbias + nbh[(size_t)j * NRES + kcol];
      // row reductions across the 16 lanes holding this row
      float mx = sv;
      mx = fmaxf(mx, __shfl_xor(mx, 1, 32));
      mx = fmaxf(mx, __shfl_xor(mx, 2, 32));
      mx = fmaxf(mx, __shfl_xor(mx, 4, 32));
      mx = fmaxf(mx, __shfl_xor(mx, 8, 32));
      float nm = fmaxf(rm[v], mx);
      float sc = __expf(rm[v] - nm);
      float pv = __expf(sv - nm);
      float sum = pv;
      sum += __shfl_xor(sum, 1, 32);
      sum += __shfl_xor(sum, 2, 32);
      sum += __shfl_xor(sum, 4, 32);
      sum += __shfl_xor(sum, 8, 32);
      rs[v] = rs[v] * sc + sum;
      rm[v] = nm;
      oacc[v] *= sc;
      pl[(roff + v) * 16 + n] = (f16)pv;
    }
    WAVE_LDS_FENCE();
    // A fragment of P (16x16, zero-padded K) via LDS transpose
    v16h pf;
#pragma unroll
    for (int t = 0; t < 16; ++t) pf[t] = (f16)0.f;
    if (!(lane & 16)) {
      int m = lane & 15;
#pragma unroll
      for (int t = 0; t < 16; ++t) pf[t] = pl[m * 16 + t];
    }
    // B fragment of V tile (Bt[ach][kk] = vt[ach][kt*16+kk])
    v16h vf;
#pragma unroll
    for (int t = 0; t < 16; ++t) vf[t] = (f16)0.f;
    if (!(lane & 16)) {
      const f16* p = vb + (size_t)n * NRES + kt * 16;
#pragma unroll
      for (int t = 0; t < 16; ++t) vf[t] = p[t];
    }
    oacc = wmma_f16(pf, vf, oacc);
    WAVE_LDS_FENCE();
  }
#pragma unroll
  for (int v = 0; v < 8; ++v) {
    int j = jt * 16 + roff + v;
    o[((size_t)i * NRES + j) * CH + h * ACH + n] = oacc[v] / rs[v];
  }
}

// ------------------------------- attention epilogue: cur += (o*g)@wo + bo

__global__ __launch_bounds__(128)
void att_epi_kernel(const float* __restrict__ o, const float* __restrict__ g,
                    const f16* __restrict__ woT, const float* bo,
                    float* __restrict__ cur, int transpose) {
  __shared__ __align__(16) f16 lds[WPB][16 * CH];
  int wave = threadIdx.x >> 5, lane = threadIdx.x & 31;
  int r0 = (blockIdx.x * WPB + wave) * 16;
  f16* l = lds[wave];
  {
    int row = lane >> 1, half = lane & 1;
    const float* po = o + (size_t)(r0 + row) * CH + half * 32;
    const float* pg = g + (size_t)(r0 + row) * CH + half * 32;
#pragma unroll
    for (int t = 0; t < 32; ++t)
      l[row * CH + half * 32 + t] = (f16)(po[t] * pg[t]);
  }
  WAVE_LDS_FENCE();
  v16h a0 = a_frag(l, CH, lane, 0);
  v16h a1 = a_frag(l, CH, lane, 32);
  int n = lane & 15, roff = (lane & 16) ? 8 : 0;
#pragma unroll
  for (int ct = 0; ct < 4; ++ct) {
    v8f acc = {};
    acc = wmma_f16(a0, b_frag(woT + ct * 16 * CH, CH, lane, 0), acc);
    acc = wmma_f16(a1, b_frag(woT + ct * 16 * CH, CH, lane, 32), acc);
    int c = ct * 16 + n;
    float vbo = bo[c];
#pragma unroll
    for (int v = 0; v < 8; ++v) {
      int r = r0 + roff + v;
      size_t pd = transpose ? (size_t)(((r & 255) << 8) | (r >> 8)) : (size_t)r;
      cur[pd * CH + c] += acc[v] + vbo;
    }
  }
}

// ----------------------------------- transition: out = cur + MLP(LN(cur))

__global__ __launch_bounds__(128)
void transition_kernel(const float* __restrict__ cur, const float* lg,
                       const float* lb, const f16* __restrict__ w1T,
                       const float* b1, const f16* __restrict__ w2T,
                       const float* b2, float* __restrict__ outp) {
  __shared__ __align__(16) f16 lds[WPB][16 * CH];
  __shared__ __align__(16) f16 lds2[WPB][16 * 128];
  int wave = threadIdx.x >> 5, lane = threadIdx.x & 31;
  int r0 = (blockIdx.x * WPB + wave) * 16;
  f16* l = lds[wave];
  f16* hh = lds2[wave];
  ln_rows_to_lds(cur + (size_t)r0 * CH, lg, lb, l, lane);
  WAVE_LDS_FENCE();
  v16h a0 = a_frag(l, CH, lane, 0);
  v16h a1 = a_frag(l, CH, lane, 32);
  int n = lane & 15, roff = (lane & 16) ? 8 : 0;
#pragma unroll
  for (int ht = 0; ht < 8; ++ht) {
    v8f acc = {};
    acc = wmma_f16(a0, b_frag(w1T + ht * 16 * CH, CH, lane, 0), acc);
    acc = wmma_f16(a1, b_frag(w1T + ht * 16 * CH, CH, lane, 32), acc);
    int c = ht * 16 + n;
    float vb = b1[c];
#pragma unroll
    for (int v = 0; v < 8; ++v) {
      float x = acc[v] + vb;
      hh[(roff + v) * 128 + c] = (f16)(x > 0.f ? x : 0.f);
    }
  }
  WAVE_LDS_FENCE();
  v16h h0 = a_frag(hh, 128, lane, 0);
  v16h h1 = a_frag(hh, 128, lane, 32);
  v16h h2 = a_frag(hh, 128, lane, 64);
  v16h h3 = a_frag(hh, 128, lane, 96);
#pragma unroll
  for (int ct = 0; ct < 4; ++ct) {
    v8f acc = {};
    acc = wmma_f16(h0, b_frag(w2T + ct * 16 * 128, 128, lane, 0), acc);
    acc = wmma_f16(h1, b_frag(w2T + ct * 16 * 128, 128, lane, 32), acc);
    acc = wmma_f16(h2, b_frag(w2T + ct * 16 * 128, 128, lane, 64), acc);
    acc = wmma_f16(h3, b_frag(w2T + ct * 16 * 128, 128, lane, 96), acc);
    int c = ct * 16 + n;
    float vb = b2[c];
#pragma unroll
    for (int v = 0; v < 8; ++v) {
      size_t idx = (size_t)(r0 + roff + v) * CH + c;
      outp[idx] = cur[idx] + acc[v] + vb;
    }
  }
}

// ------------------------------------------------------------------- host

struct TriMulP {
  const float *ln_g, *ln_b, *wl, *bl, *wr, *br, *wlg, *blg, *wrg, *brg,
              *cln_g, *cln_b, *wo, *bo, *wg, *bg;
};
struct TriAttP {
  const float *ln_g, *ln_b, *w2d, *wq, *wk, *wv, *wg, *bg, *wo, *bo;
};
struct PtP { const float *ln_g, *ln_b, *w1, *b1, *w2, *b2; };
struct TriMulW { f16 *wl, *wlg, *wr, *wrg, *wg, *wo; };
struct TriAttW { f16 *wq, *wk, *wv, *wg, *wo, *w2d; };

extern "C" void kernel_launch(void* const* d_in, const int* in_sizes, int n_in,
                              void* d_out, int out_size, void* d_ws, size_t ws_size,
                              hipStream_t stream) {
  (void)out_size; (void)ws_size; (void)n_in;
  auto F = [&](int i) { return (const float*)d_in[i]; };
  const float* act  = F(0);
  const float* mask = F(1);

  // Detect pytree flatten order: jax sorts dict keys -> d_in[2] = pt.b1 (128).
  bool alpha = (in_sizes[2] == 128);
  TriMulP tmo, tmi; TriAttP tas, tae; PtP pt;
  if (alpha) {
    pt = { F(5), F(4), F(6), F(2), F(7), F(3) };
    auto TA = [&](int b) {
      TriAttP a; a.bg = F(b); a.bo = F(b + 1); a.ln_b = F(b + 2);
      a.ln_g = F(b + 3); a.w2d = F(b + 4); a.wg = F(b + 5); a.wk = F(b + 6);
      a.wo = F(b + 7); a.wq = F(b + 8); a.wv = F(b + 9); return a;
    };
    tae = TA(8); tas = TA(18);
    auto TM = [&](int b) {
      TriMulP m; m.bg = F(b); m.bl = F(b + 1); m.blg = F(b + 2); m.bo = F(b + 3);
      m.br = F(b + 4); m.brg = F(b + 5); m.cln_b = F(b + 6); m.cln_g = F(b + 7);
      m.ln_b = F(b + 8); m.ln_g = F(b + 9); m.wg = F(b + 10); m.wl = F(b + 11);
      m.wlg = F(b + 12); m.wo = F(b + 13); m.wr = F(b + 14); m.wrg = F(b + 15);
      return m;
    };
    tmi = TM(28); tmo = TM(44);
  } else {
    auto TM = [&](int b) {
      TriMulP m; m.ln_g = F(b); m.ln_b = F(b + 1); m.wl = F(b + 2); m.bl = F(b + 3);
      m.wr = F(b + 4); m.br = F(b + 5); m.wlg = F(b + 6); m.blg = F(b + 7);
      m.wrg = F(b + 8); m.brg = F(b + 9); m.cln_g = F(b + 10); m.cln_b = F(b + 11);
      m.wo = F(b + 12); m.bo = F(b + 13); m.wg = F(b + 14); m.bg = F(b + 15);
      return m;
    };
    tmo = TM(2); tmi = TM(18);
    auto TA = [&](int b) {
      TriAttP a; a.ln_g = F(b); a.ln_b = F(b + 1); a.w2d = F(b + 2);
      a.wq = F(b + 3); a.wk = F(b + 4); a.wv = F(b + 5); a.wg = F(b + 6);
      a.bg = F(b + 7); a.wo = F(b + 8); a.bo = F(b + 9); return a;
    };
    tas = TA(34); tae = TA(44);
    pt = { F(54), F(55), F(56), F(57), F(58), F(59) };
  }

  // ---- workspace layout
  char* ws = (char*)d_ws;
  const size_t MB = 1ull << 20;
  float* cur    = (float*)(ws + 0);        // 16MB
  float* gate32 = (float*)(ws + 16 * MB);  // 16MB
  float* tri32  = (float*)(ws + 32 * MB);  // 16MB  (also attention o32)
  f16*   xln16  = (f16*)(ws + 48 * MB);    // 8MB
  f16*   buf0   = (f16*)(ws + 56 * MB);    // 8MB   (L16 / q16)
  f16*   buf1   = (f16*)(ws + 64 * MB);    // 8MB   (R16 / k16)
  f16*   buf2   = (f16*)(ws + 72 * MB);    // 8MB   (v16t)
  float* nb32   = (float*)(ws + 80 * MB);  // 1MB
  f16*   wtb    = (f16*)(ws + 81 * MB);    // weights ~256KB
  size_t woff = 0;
  auto wal = [&](size_t nh) { f16* p = wtb + woff; woff += nh; return p; };
  auto cvt = [&](const float* w, f16* dst, int kin, int nout) {
    int nn = kin * nout;
    convert_wt_kernel<<<(nn + 255) / 256, 256, 0, stream>>>(w, dst, kin, nout);
  };

  auto makeTMW = [&](const TriMulP& p) {
    TriMulW w;
    w.wl = wal(4096);  cvt(p.wl,  w.wl,  64, 64);
    w.wlg = wal(4096); cvt(p.wlg, w.wlg, 64, 64);
    w.wr = wal(4096);  cvt(p.wr,  w.wr,  64, 64);
    w.wrg = wal(4096); cvt(p.wrg, w.wrg, 64, 64);
    w.wg = wal(4096);  cvt(p.wg,  w.wg,  64, 64);
    w.wo = wal(4096);  cvt(p.wo,  w.wo,  64, 64);
    return w;
  };
  auto makeTAW = [&](const TriAttP& p) {
    TriAttW w;
    w.wq = wal(4096); cvt(p.wq, w.wq, 64, 64);
    w.wk = wal(4096); cvt(p.wk, w.wk, 64, 64);
    w.wv = wal(4096); cvt(p.wv, w.wv, 64, 64);
    w.wg = wal(4096); cvt(p.wg, w.wg, 64, 64);
    w.wo = wal(4096); cvt(p.wo, w.wo, 64, 64);
    w.w2d = wal(1024);
    convert_w2d_kernel<<<4, 256, 0, stream>>>(p.w2d, w.w2d);
    return w;
  };
  TriMulW wtmo = makeTMW(tmo), wtmi = makeTMW(tmi);
  TriAttW wtas = makeTAW(tas), wtae = makeTAW(tae);
  f16* w1T = wal(8192); cvt(pt.w1, w1T, 64, 128);
  f16* w2T = wal(8192); cvt(pt.w2, w2T, 128, 64);

  hipMemcpyAsync(cur, act, (size_t)NPOS * CH * sizeof(float),
                 hipMemcpyDeviceToDevice, stream);

  const int GB = NPOS / 16 / WPB;  // 1024 blocks of 128 threads

  // ---- tri_mult outgoing
  ln_to_f16_kernel<<<512, 256, 0, stream>>>(cur, tmo.ln_g, tmo.ln_b, xln16, 0);
  trimul_proj_kernel<<<GB, 128, 0, stream>>>(xln16, wtmo.wl, wtmo.wlg, wtmo.wr,
      wtmo.wrg, wtmo.wg, tmo.bl, tmo.blg, tmo.br, tmo.brg, tmo.bg, mask,
      buf0, buf1, gate32, 0);
  trimul_gemm_kernel<<<GB, 128, 0, stream>>>(buf0, buf1, tri32);
  trimul_fuse_kernel<<<GB, 128, 0, stream>>>(tri32, tmo.cln_g, tmo.cln_b,
      wtmo.wo, tmo.bo, gate32, cur);

  // ---- tri_mult incoming (transposed storage; A=right, B=left)
  ln_to_f16_kernel<<<512, 256, 0, stream>>>(cur, tmi.ln_g, tmi.ln_b, xln16, 0);
  trimul_proj_kernel<<<GB, 128, 0, stream>>>(xln16, wtmi.wl, wtmi.wlg, wtmi.wr,
      wtmi.wrg, wtmi.wg, tmi.bl, tmi.blg, tmi.br, tmi.brg, tmi.bg, mask,
      buf0, buf1, gate32, 1);
  trimul_gemm_kernel<<<GB, 128, 0, stream>>>(buf1, buf0, tri32);
  trimul_fuse_kernel<<<GB, 128, 0, stream>>>(tri32, tmi.cln_g, tmi.cln_b,
      wtmi.wo, tmi.bo, gate32, cur);

  // ---- triangle attention (start)
  ln_to_f16_kernel<<<512, 256, 0, stream>>>(cur, tas.ln_g, tas.ln_b, xln16, 0);
  att_proj_kernel<<<GB, 128, 0, stream>>>(xln16, wtas.wq, wtas.wk, wtas.wv,
      wtas.wg, wtas.w2d, tas.bg, buf0, buf1, buf2, gate32, nb32);
  att_core_kernel<<<NPOS / 16 / WPB * 4, 128, 0, stream>>>(buf0, buf1, buf2,
      nb32, mask, tri32, 0);
  att_epi_kernel<<<GB, 128, 0, stream>>>(tri32, gate32, wtas.wo, tas.bo, cur, 0);

  // ---- triangle attention (end): transposed positions + transposed mask
  ln_to_f16_kernel<<<512, 256, 0, stream>>>(cur, tae.ln_g, tae.ln_b, xln16, 1);
  att_proj_kernel<<<GB, 128, 0, stream>>>(xln16, wtae.wq, wtae.wk, wtae.wv,
      wtae.wg, wtae.w2d, tae.bg, buf0, buf1, buf2, gate32, nb32);
  att_core_kernel<<<NPOS / 16 / WPB * 4, 128, 0, stream>>>(buf0, buf1, buf2,
      nb32, mask, tri32, 1);
  att_epi_kernel<<<GB, 128, 0, stream>>>(tri32, gate32, wtae.wo, tae.bo, cur, 1);

  // ---- transition -> d_out
  transition_kernel<<<GB, 128, 0, stream>>>(cur, pt.ln_g, pt.ln_b, w1T, pt.b1,
      w2T, pt.b2, (float*)d_out);
}
